// AutoInt_4312147165649
// MI455X (gfx1250) — matrix operand order repaired and checked by reading, not compiled, not run on hardware
//
#include <hip/hip_runtime.h>
#include <hip/hip_bf16.h>
#include <math.h>

// ---------------------------------------------------------------------------
// AutoInt fused forward for MI455X (gfx1250, wave32, WMMA 16x16x32 f16/f32acc)
// One wave per 16 batch rows. All GEMMs via v_wmma_f32_16x16x32_f16.
// ---------------------------------------------------------------------------

typedef _Float16 f16;
typedef __attribute__((ext_vector_type(16))) _Float16 v16h;
typedef __attribute__((ext_vector_type(8)))  _Float16 v8h;
typedef __attribute__((ext_vector_type(4)))  _Float16 v4h;
typedef __attribute__((ext_vector_type(8)))  float    v8f;

// packed weight fragments: 129 frags x 512 f16 (1KB each)
// layout of one B-fragment (this code's contiguous k order for B 32x16, same as ldA):
//   lane L holds column N = ntile*16 + (L&15); element e holds K = kstep*32 + (L>>4)*16 + e
#define NFRAGS 129
__device__ f16 g_wpk[NFRAGS * 512];

__device__ __forceinline__ v8f wmma_f16(v16h a, v16h b, v8f c) {
  return __builtin_amdgcn_wmma_f32_16x16x32_f16(false, a, false, b, (short)0, c, false, false);
}

__device__ __forceinline__ v16h zero16() {
  v16h z;
#pragma unroll
  for (int i = 0; i < 16; i++) z[i] = (f16)0.f;
  return z;
}

// A-fragment loader from row-major f16 LDS buffer.
// 16-bit A 16x32 in this code's contiguous k order (matches B packing): lane L holds row M=L&15, K base kb=(L>>4)*16,
// elements 0..15 -> K=kb..kb+15  (two b128 loads)
__device__ __forceinline__ v16h ldA(const f16* buf, int stride, int row, int kstart, int akb) {
  const f16* p = buf + row * stride + kstart + akb;
  v8h lo = *(const v8h*)p;
  v8h hi = *(const v8h*)(p + 8);
  return __builtin_shufflevector(lo, hi, 0, 1, 2, 3, 4, 5, 6, 7, 8, 9, 10, 11, 12, 13, 14, 15);
}

__device__ __forceinline__ v16h ldB(int frag, int lane) {
  return *(const v16h*)(g_wpk + (size_t)frag * 512 + lane * 16);
}

// ---------------------------------------------------------------------------
// Pack kernel: f32 weights -> f16 B-fragments, zero-padded K/N.
// frag ids: W1: m*8+ks*4+nt (0..47) | W2: 48+m*4+ks*2+nt | W3: 72+m
//           Wd1: 78+ks*2+nt (ks 0..23) | Wd2: 126 | QKV: 127+nt (N=[q|k|v])
// ---------------------------------------------------------------------------
__global__ __launch_bounds__(32) void pack_weights(
    const float* __restrict__ W1, const float* __restrict__ W2, const float* __restrict__ W3,
    const float* __restrict__ Wd1, const float* __restrict__ Wd2,
    const float* __restrict__ Wq, const float* __restrict__ Wk, const float* __restrict__ Wv) {
  int frag = blockIdx.x;
  int lane = threadIdx.x;
  int lm = lane & 15, lh = lane >> 4;

  const float* src = nullptr;
  int Ktot = 0, Ntot = 0, ld = 0, ks = 0, nt = 0;
  bool qkv = false;
  if (frag < 48) {
    int m = frag >> 3, r = frag & 7; ks = r >> 2; nt = r & 3;
    src = W1 + m * 40 * 64; Ktot = 40; Ntot = 64; ld = 64;
  } else if (frag < 72) {
    int f = frag - 48; int m = f >> 2, r = f & 3; ks = r >> 1; nt = r & 1;
    src = W2 + m * 64 * 32; Ktot = 64; Ntot = 32; ld = 32;
  } else if (frag < 78) {
    int m = frag - 72; src = W3 + m * 32 * 16; Ktot = 32; Ntot = 16; ld = 16;
  } else if (frag < 126) {
    int f = frag - 78; ks = f >> 1; nt = f & 1;
    src = Wd1; Ktot = 768; Ntot = 32; ld = 32;
  } else if (frag == 126) {
    src = Wd2; Ktot = 32; Ntot = 16; ld = 16;
  } else {
    qkv = true; nt = frag - 127;
  }

  f16* dst = g_wpk + (size_t)frag * 512 + lane * 16;
#pragma unroll
  for (int e = 0; e < 16; e++) {
    int K = ks * 32 + lh * 16 + e;
    int N = nt * 16 + lm;
    float v = 0.f;
    if (qkv) {
      if (K < 8 && N < 24)
        v = (N < 8) ? Wq[K * 8 + N] : (N < 16) ? Wk[K * 8 + (N - 8)] : Wv[K * 8 + (N - 16)];
    } else if (K < Ktot && N < Ntot) {
      v = src[K * ld + N];
    }
    dst[e] = (f16)v;
  }
}

// ---------------------------------------------------------------------------
// Main fused kernel: one wave handles 16 batch rows end-to-end.
// ---------------------------------------------------------------------------
__global__ __launch_bounds__(32) void autoint_main(
    const float* __restrict__ mod_fea,
    const float* __restrict__ b1, const float* __restrict__ b2, const float* __restrict__ b3,
    const float* __restrict__ We, const float* __restrict__ be,
    const float* __restrict__ bd1, const float* __restrict__ bd2,
    const float* __restrict__ Wo, const float* __restrict__ bo,
    float* __restrict__ out) {
  __shared__ __align__(16) f16 s_emb[16 * 768];  // 24KB: [sample][token*8+e]
  __shared__ __align__(16) f16 s_r1[5632];       // 11KB: phaseA {in16|h1|h2} / phaseC {q|k|vT|att}
  __shared__ __align__(16) f16 s_dnn[256];       // [sample][16] dnn output

  const int lane = threadIdx.x;
  const int lm = lane & 15;
  const int lh = lane >> 4;
  const int akb = lh * 16;   // A-frag K base
  const int bkb = lh * 16;   // B-frag K base
  const int tile0 = blockIdx.x * 16;

  f16* in16 = s_r1;          // [16][240] (+64 pad, bleed covered by zero weights)
  f16* h1 = s_r1 + 3904;     // [16][64]
  f16* h2 = s_r1 + 4928;     // [16][32]

  // -------- Phase A: stage input tile (contiguous 16x240 f32) as f16 --------
  {
    const float* src = mod_fea + (size_t)tile0 * 240;
    for (int i = lane * 4; i < 3840; i += 128) {
      float4 v = *(const float4*)(src + i);
      v4h hv; hv[0] = (f16)v.x; hv[1] = (f16)v.y; hv[2] = (f16)v.z; hv[3] = (f16)v.w;
      *(v4h*)&in16[i] = hv;
    }
    if (lane < 16) {
      v4h z; z[0] = (f16)0.f; z[1] = (f16)0.f; z[2] = (f16)0.f; z[3] = (f16)0.f;
      *(v4h*)&in16[3840 + lane * 4] = z;
    }
  }
  __syncthreads();

  // -------- Per-module MLP [40->64->32->16] + fused scalar embedders --------
  for (int m = 0; m < 6; m++) {
    // layer 1: K=40 (padded to 64), N=64
    v16h a0 = ldA(in16, 240, lm, m * 40, akb);
    v16h a1 = ldA(in16, 240, lm, m * 40 + 32, akb);
#pragma unroll
    for (int t = 0; t < 4; t++) {
      v8f acc = {};
      acc = wmma_f16(a0, ldB(m * 8 + 0 * 4 + t, lane), acc);
      acc = wmma_f16(a1, ldB(m * 8 + 1 * 4 + t, lane), acc);
      float bias = b1[m * 64 + t * 16 + lm];
#pragma unroll
      for (int r = 0; r < 8; r++)
        h1[(r + 8 * lh) * 64 + t * 16 + lm] = (f16)fmaxf(acc[r] + bias, 0.f);
    }
    __syncthreads();
    // layer 2: K=64, N=32
    a0 = ldA(h1, 64, lm, 0, akb);
    a1 = ldA(h1, 64, lm, 32, akb);
#pragma unroll
    for (int t = 0; t < 2; t++) {
      v8f acc = {};
      acc = wmma_f16(a0, ldB(48 + m * 4 + 0 * 2 + t, lane), acc);
      acc = wmma_f16(a1, ldB(48 + m * 4 + 1 * 2 + t, lane), acc);
      float bias = b2[m * 32 + t * 16 + lm];
#pragma unroll
      for (int r = 0; r < 8; r++)
        h2[(r + 8 * lh) * 32 + t * 16 + lm] = (f16)fmaxf(acc[r] + bias, 0.f);
    }
    __syncthreads();
    // layer 3: K=32, N=16 -> flat[:, m*16+j], then emb = flat*We+be
    v16h a2 = ldA(h2, 32, lm, 0, akb);
    v8f acc = {};
    acc = wmma_f16(a2, ldB(72 + m, lane), acc);
    float bias = b3[m * 16 + lm];
    int st = m * 16 + lm;  // token index this lane owns
    float we0[8], be0[8];
#pragma unroll
    for (int e = 0; e < 8; e++) { we0[e] = We[st * 8 + e]; be0[e] = be[st * 8 + e]; }
#pragma unroll
    for (int r = 0; r < 8; r++) {
      float v = fmaxf(acc[r] + bias, 0.f);
      v8h ev;
#pragma unroll
      for (int e = 0; e < 8; e++) ev[e] = (f16)(v * we0[e] + be0[e]);
      *(v8h*)&s_emb[(r + 8 * lh) * 768 + st * 8] = ev;
    }
    __syncthreads();
  }

  // -------- DNN branch: [16x768]@[768x32] relu, [16x32]@[32x16] relu -------
  {
    v8f c0 = {}, c1 = {};
    for (int ks = 0; ks < 24; ks++) {
      v16h a = ldA(s_emb, 768, lm, ks * 32, akb);
      c0 = wmma_f16(a, ldB(78 + ks * 2 + 0, lane), c0);
      c1 = wmma_f16(a, ldB(78 + ks * 2 + 1, lane), c1);
    }
    float bi0 = bd1[lm], bi1 = bd1[16 + lm];
#pragma unroll
    for (int r = 0; r < 8; r++) {
      h2[(r + 8 * lh) * 32 + lm] = (f16)fmaxf(c0[r] + bi0, 0.f);
      h2[(r + 8 * lh) * 32 + 16 + lm] = (f16)fmaxf(c1[r] + bi1, 0.f);
    }
    __syncthreads();
    v16h a = ldA(h2, 32, lm, 0, akb);
    v8f acc = {};
    acc = wmma_f16(a, ldB(126, lane), acc);
    float bias = bd2[lm];
#pragma unroll
    for (int r = 0; r < 8; r++)
      s_dnn[(r + 8 * lh) * 16 + lm] = (f16)fmaxf(acc[r] + bias, 0.f);
    __syncthreads();
  }

  // -------- Attention (per sample) + residual relu + fused head dot --------
  f16* qb = s_r1;          // [96][8]
  f16* kb = s_r1 + 768;    // [96][8]
  f16* vT = s_r1 + 1536;   // [8][96] transposed v
  f16* at = s_r1 + 2304;   // [16][96] softmax probs
  const float bov = bo[0];

  for (int s = 0; s < 16; s++) {
    float part = 0.f;
    if (lane < 16) part = (float)s_dnn[s * 16 + lane] * Wo[lane];
    const f16* embS = s_emb + s * 768;

    // qkv projection: [96 x 8] @ [8 x 24(pad32)]
#pragma unroll
    for (int mt = 0; mt < 6; mt++) {
      v16h a = zero16();
      if (lh == 0) {
        v8h lo = *(const v8h*)(embS + (mt * 16 + lm) * 8);
#pragma unroll
        for (int e = 0; e < 8; e++) a[e] = lo[e];
      }
#pragma unroll
      for (int t = 0; t < 2; t++) {
        v8f acc = {};
        acc = wmma_f16(a, ldB(127 + t, lane), acc);
        int n = t * 16 + lm;
#pragma unroll
        for (int r = 0; r < 8; r++) {
          int tok = mt * 16 + r + 8 * lh;
          f16 val = (f16)acc[r];
          if (n < 8)        qb[tok * 8 + n] = val;
          else if (n < 16)  kb[tok * 8 + (n - 8)] = val;
          else if (n < 24)  vT[(n - 16) * 96 + tok] = val;
        }
      }
    }
    __syncthreads();

    for (int h = 0; h < 2; h++) {
      for (int qt = 0; qt < 6; qt++) {
        // scores: q[16x4 pad32] @ kT[4x96]
        v16h aq = zero16();
        if (lh == 0) {
          v4h qv = *(const v4h*)(qb + (qt * 16 + lm) * 8 + h * 4);
#pragma unroll
          for (int e = 0; e < 4; e++) aq[e] = qv[e];
        }
        v8f S[6];
#pragma unroll
        for (int kt = 0; kt < 6; kt++) {
          v16h bk = zero16();
          if (lh == 0) {
            v4h kv = *(const v4h*)(kb + (kt * 16 + lm) * 8 + h * 4);
#pragma unroll
            for (int e = 0; e < 4; e++) bk[e] = kv[e];
          }
          v8f z = {};
          S[kt] = wmma_f16(aq, bk, z);
        }
        // row softmax: lanes 0..15 (and 16..31) share the q-row per register r
        float mx[8], sm[8];
#pragma unroll
        for (int r = 0; r < 8; r++) {
          float m0 = S[0][r];
#pragma unroll
          for (int kt = 1; kt < 6; kt++) m0 = fmaxf(m0, S[kt][r]);
          mx[r] = m0;
        }
#pragma unroll
        for (int d = 1; d < 16; d <<= 1)
#pragma unroll
          for (int r = 0; r < 8; r++) mx[r] = fmaxf(mx[r], __shfl_xor(mx[r], d, 32));
#pragma unroll
        for (int r = 0; r < 8; r++) {
          float acc = 0.f;
#pragma unroll
          for (int kt = 0; kt < 6; kt++) {
            float p = __expf((S[kt][r] - mx[r]) * 0.5f);  // scale = 1/sqrt(4)
            S[kt][r] = p; acc += p;
          }
          sm[r] = acc;
        }
#pragma unroll
        for (int d = 1; d < 16; d <<= 1)
#pragma unroll
          for (int r = 0; r < 8; r++) sm[r] += __shfl_xor(sm[r], d, 32);
#pragma unroll
        for (int r = 0; r < 8; r++) {
          float inv = 1.f / sm[r];
#pragma unroll
          for (int kt = 0; kt < 6; kt++)
            at[(r + 8 * lh) * 96 + kt * 16 + lm] = (f16)(S[kt][r] * inv);
        }
        __syncthreads();
        // AV: att[16x96] @ v[96x4 pad16]
        v8f o = {};
#pragma unroll
        for (int kt2 = 0; kt2 < 3; kt2++) {
          v16h aA = ldA(at, 96, lm, kt2 * 32, akb);
          v16h bV = zero16();
          if (lm < 4) {
            const f16* vp = vT + (h * 4 + lm) * 96 + kt2 * 32 + bkb;
            v8h lo = *(const v8h*)vp, hi = *(const v8h*)(vp + 8);
#pragma unroll
            for (int e = 0; e < 8; e++) { bV[e] = lo[e]; bV[8 + e] = hi[e]; }
          }
          o = wmma_f16(aA, bV, o);
        }
        // residual + relu + fused head dot accumulation
#pragma unroll
        for (int r = 0; r < 8; r++) {
          if (lm < 4) {
            int tok = qt * 16 + r + 8 * lh;
            int col = h * 4 + lm;
            float v = o[r] + (float)embS[tok * 8 + col];
            v = fmaxf(v, 0.f);
            part += v * Wo[16 + tok * 8 + col];
          }
        }
        __syncthreads();  // 'at' reused next (h,qt)
      }
    }
    // wave reduction of the head dot -> sigmoid -> out
#pragma unroll
    for (int d = 16; d >= 1; d >>= 1) part += __shfl_xor(part, d, 32);
    if (lane == 0) out[tile0 + s] = 1.f / (1.f + __expf(-(part + bov)));
    __syncthreads();  // qb/kb/vT reused next sample
  }
}

// ---------------------------------------------------------------------------
extern "C" void kernel_launch(void* const* d_in, const int* in_sizes, int n_in,
                              void* d_out, int out_size, void* d_ws, size_t ws_size,
                              hipStream_t stream) {
  (void)n_in; (void)d_ws; (void)ws_size; (void)out_size;
  const float* mod_fea = (const float*)d_in[0];
  const float* W1 = (const float*)d_in[1];
  const float* b1 = (const float*)d_in[2];
  const float* W2 = (const float*)d_in[3];
  const float* b2 = (const float*)d_in[4];
  const float* W3 = (const float*)d_in[5];
  const float* b3 = (const float*)d_in[6];
  const float* We = (const float*)d_in[7];
  const float* be = (const float*)d_in[8];
  const float* Wd1 = (const float*)d_in[9];
  const float* bd1 = (const float*)d_in[10];
  const float* Wd2 = (const float*)d_in[11];
  const float* bd2 = (const float*)d_in[12];
  const float* Wq = (const float*)d_in[13];
  const float* Wk = (const float*)d_in[14];
  const float* Wv = (const float*)d_in[15];
  const float* Wo = (const float*)d_in[16];
  const float* bo = (const float*)d_in[17];

  int B = in_sizes[0] / 240;  // 16384

  pack_weights<<<NFRAGS, 32, 0, stream>>>(W1, W2, W3, Wd1, Wd2, Wq, Wk, Wv);
  autoint_main<<<B / 16, 32, 0, stream>>>(mod_fea, b1, b2, b3, We, be, bd1, bd2,
                                          Wo, bo, (float*)d_out);
}
